// CosformerAttention_4037269258963
// MI455X (gfx1250) — matrix operand order repaired and checked by reading, not compiled, not run on hardware
//
#include <hip/hip_runtime.h>
#include <hip/hip_bf16.h>

// ---------------------------------------------------------------------------
// Cosformer attention for MI455X (gfx1250, wave32, WMMA + TDM).
//   L=4096, N=2, E=1024, H=16, D=64  ->  M = L*N = 8192 rows
// GEMMs (69 GFLOP) dominate; bf16 WMMA w/ f32 accumulate, fed from LDS by the
// Tensor Data Mover (double-buffered, TENSORcnt-tracked, padded LDS rows).
// Linear-attention middle (4 GFLOP) in fp32 VALU with LDS staging.
// ---------------------------------------------------------------------------

#define LSEQ   4096
#define NBATCH 2
#define EDIM   1024
#define NHEADS 16
#define DHEAD  64
#define MROWS  (LSEQ * NBATCH)     // 8192
#define NH_TOT (NBATCH * NHEADS)   // 32 heads
#define TWOD   (2 * DHEAD)         // 128

typedef __attribute__((ext_vector_type(16))) __bf16 v16bf;
typedef __attribute__((ext_vector_type(8)))  float  v8f;
typedef __attribute__((ext_vector_type(4)))  unsigned int v4u;
typedef __attribute__((ext_vector_type(4)))  int    v4i;
typedef __attribute__((ext_vector_type(8)))  int    v8i;

struct Bf16Frag { float4 lo, hi; };   // 32 bytes == one 16-bf16 lane fragment

__device__ __forceinline__ float bf2f(unsigned short u) {
    unsigned int x = ((unsigned int)u) << 16;
    return __builtin_bit_cast(float, x);
}
__device__ __forceinline__ unsigned short f2bf(float f) {
    unsigned int u = __builtin_bit_cast(unsigned int, f);
    u += 0x7FFFu + ((u >> 16) & 1u);          // round-to-nearest-even
    return (unsigned short)(u >> 16);
}

// TDM 2D tile load: one descriptor pair (group0 runtime, group1 shared).
__device__ __forceinline__ void tdm_load_2d(unsigned lds_addr,
                                            unsigned long long gaddr,
                                            v8i g1) {
    v4u g0;
    g0[0] = 1u;                                   // count=1, user mode, no gather
    g0[1] = lds_addr;                             // LDS byte address
    g0[2] = (unsigned)(gaddr & 0xFFFFFFFFu);      // global_addr[31:0]
    g0[3] = (unsigned)((gaddr >> 32) & 0x1FFFFFFu) | 0x80000000u; // [56:32] | type=2
    v4i z4 = {0, 0, 0, 0};
#if __has_include(<hip/amd_detail/amd_gfx1250_TDM.h>)
    v8i z8 = {0, 0, 0, 0, 0, 0, 0, 0};            // therock clang-23: 6-arg form
    __builtin_amdgcn_tensor_load_to_lds(g0, g1, z4, z4, z8, 0);
#else
    __builtin_amdgcn_tensor_load_to_lds(g0, g1, z4, z4, 0);      // ROCm 7.2: 5-arg
#endif
}

// ---------------------------------------------------------------------------
// fp32 -> bf16 bulk convert (vectorized: float4 in, 4x u16 out)
// ---------------------------------------------------------------------------
__global__ __launch_bounds__(256) void f32_to_bf16_kernel(
    const float* __restrict__ src, unsigned short* __restrict__ dst, int n)
{
    int i = (blockIdx.x * blockDim.x + threadIdx.x) * 4;
    if (i + 3 < n) {
        float4 v = *(const float4*)(src + i);
        ushort4 o;
        o.x = f2bf(v.x); o.y = f2bf(v.y); o.z = f2bf(v.z); o.w = f2bf(v.w);
        *(ushort4*)(dst + i) = o;
    }
}

// ---------------------------------------------------------------------------
// C[M,N] = act(A[M,K](bf16) @ B[N,K](bf16)^T)    (both K-contiguous row-major)
// Block = 8 waves -> 128x128 tile; wave -> 32x64 (2 A-frags x 4 B-frags,
// 8 WMMA / K-step). A/B K-tiles (128x32 bf16 each) staged in LDS by the TDM,
// double-buffered; LDS rows padded to 80 B by the TDM for conflict-free
// ds_load_b128 fragment reads. MODE: 0 = bf16 out, 1 = bf16+relu, 2 = f32.
// ---------------------------------------------------------------------------
#define BK 32
#define ROWB 80                          // 64 B data + 16 B TDM pad
#define TILE_BYTES (128 * ROWB)          // 10240 per tile

template <int MODE>
__global__ __launch_bounds__(256) void gemm_wmma_tdm(
    const unsigned short* __restrict__ A,
    const unsigned short* __restrict__ B,
    void* __restrict__ Cv, int M, int Ncols, int K)
{
    __shared__ unsigned char lds[2][2 * TILE_BYTES];   // [buf][Atile | Btile]

    const int tid    = threadIdx.x;
    const int lane   = tid & 31;
    const int wave   = tid >> 5;
    const int laneHi = lane >> 4;          // 0 | 1 (K-half select)
    const int laneLo = lane & 15;          // M row / N col within 16x16 tile
    const int waveM  = wave & 3;           // 4 M-tiles of 32 rows
    const int waveN  = wave >> 2;          // 2 N-tiles of 64 cols
    const int mBase  = blockIdx.x * 128;
    const int nBase  = blockIdx.y * 128;

    // Shared TDM group-1 descriptor: 2-byte elems, tile 32(K) x 128(rows),
    // tensor dims 2^24 (no OOB), row stride = K, LDS pad 4 dw / 16 dw.
    v8i g1;
    g1[0] = (1 << 16) | (1 << 20) | (3 << 22) | (3 << 25); // dsz=2B, pad on, 16dw/4dw
    g1[1] = 0;                       // atomic barrier off, tensor_dim0 lo16 = 0
    g1[2] = 0x100;                   // tensor_dim0 = 1<<24 ; tensor_dim1 lo16 = 0
    g1[3] = 0x100 | (BK << 16);      // tensor_dim1 = 1<<24 ; tile_dim0 = 32
    g1[4] = 128;                     // tile_dim1 = 128 rows ; tile_dim2 = 0
    g1[5] = K;                       // tensor_dim0_stride
    g1[6] = 0;
    g1[7] = 0;

    const unsigned ldsBase = (unsigned)(size_t)&lds[0][0];
    const unsigned long long gA = (unsigned long long)(A + (size_t)mBase * K);
    const unsigned long long gB = (unsigned long long)(B + (size_t)nBase * K);
    const int NK = K / BK;

    v8f acc[2][4] = {};

    if (wave == 0) {   // prologue: fill buffer 0
        tdm_load_2d(ldsBase,              gA, g1);
        tdm_load_2d(ldsBase + TILE_BYTES, gB, g1);
    }

    for (int kt = 0; kt < NK; ++kt) {
        const int cur = kt & 1;
        if (wave == 0) {
            if (kt + 1 < NK) {   // issue next pair into the other buffer
                const unsigned long long koff = (unsigned long long)(kt + 1) * (BK * 2);
                const unsigned dst = ldsBase + (cur ^ 1) * (2 * TILE_BYTES);
                tdm_load_2d(dst,              gA + koff, g1);
                tdm_load_2d(dst + TILE_BYTES, gB + koff, g1);
                __builtin_amdgcn_s_wait_tensorcnt(2);  // oldest pair (cur) done
            } else {
                __builtin_amdgcn_s_wait_tensorcnt(0);  // last pair done
            }
        }
        __syncthreads();   // publish buf[cur] to all waves

        const unsigned char* aT = &lds[cur][0];
        const unsigned char* bT = aT + TILE_BYTES;

        v16bf afrag[2];
#pragma unroll
        for (int i = 0; i < 2; ++i) {
            const unsigned char* r =
                aT + (size_t)(waveM * 32 + i * 16 + laneLo) * ROWB + laneHi * 16;
            Bf16Frag fa;
            fa.lo = *(const float4*)(r);        // K chunk {0..7} / {8..15}
            fa.hi = *(const float4*)(r + 32);   // K chunk {16..23} / {24..31}
            afrag[i] = __builtin_bit_cast(v16bf, fa);
        }
        v16bf bfrag[4];
#pragma unroll
        for (int f = 0; f < 4; ++f) {
            const unsigned char* r =
                bT + (size_t)(waveN * 64 + f * 16 + laneLo) * ROWB + laneHi * 16;
            Bf16Frag fb;
            fb.lo = *(const float4*)(r);        // K chunk {0..7} / {8..15}
            fb.hi = *(const float4*)(r + 32);   // K chunk {16..23} / {24..31}
            bfrag[f] = __builtin_bit_cast(v16bf, fb);
        }

#pragma unroll
        for (int i = 0; i < 2; ++i)
#pragma unroll
            for (int f = 0; f < 4; ++f)
                acc[i][f] = __builtin_amdgcn_wmma_f32_16x16x32_bf16(
                    false, afrag[i], false, bfrag[f], (short)0, acc[i][f], false, false);

        __syncthreads();   // all waves done reading buf[cur] before reuse
    }

    // C/D layout: lane -> N = laneLo (+frag*16), VGPR v -> M = v + laneHi*8
#pragma unroll
    for (int i = 0; i < 2; ++i) {
#pragma unroll
        for (int v = 0; v < 8; ++v) {
            const size_t row = (size_t)(mBase + waveM * 32 + i * 16 + v + laneHi * 8);
            const int    col = nBase + waveN * 64 + laneLo;
#pragma unroll
            for (int f = 0; f < 4; ++f) {
                const size_t idx = row * (size_t)Ncols + (size_t)(col + f * 16);
                const float  x  = acc[i][f][v];
                if (MODE == 2) {
                    ((float*)Cv)[idx] = x;
                } else {
                    float y = (MODE == 1) ? fmaxf(x, 0.0f) : x;
                    ((unsigned short*)Cv)[idx] = f2bf(y);
                }
            }
        }
    }
}

// ---------------------------------------------------------------------------
// Per-head kv = k_^T @ v  (128x64) and ksum = sum_l k_  (128,)
// One block per head; LDS stages the RoPE+cosFormer-transformed k row (128)
// and the v row (64) per l; each thread owns 32 accumulator entries.
// ---------------------------------------------------------------------------
__global__ __launch_bounds__(256) void cosformer_kv_kernel(
    const unsigned short* __restrict__ Kbf,
    const unsigned short* __restrict__ Vbf,
    const float* __restrict__ cosT,   // (L, D)
    const float* __restrict__ sinT,   // (L, D)
    float* __restrict__ kvW,          // (NH, 128, 64)
    float* __restrict__ ksumW)        // (NH, 128)
{
    const int nh      = blockIdx.x;
    const int n_batch = nh / NHEADS;
    const int h       = nh % NHEADS;
    const int colBase = h * DHEAD;

    __shared__ float kbuf[TWOD];
    __shared__ float vbuf[DHEAD];

    const int t    = threadIdx.x;
    const int d    = t & (TWOD - 1);   // 0..127 accumulator row
    const int half = t >> 7;           // 0 | 1 -> m-range 0..31 / 32..63
    const int m0   = half * 32;

    float acc[32];
#pragma unroll
    for (int j = 0; j < 32; ++j) acc[j] = 0.0f;
    float ksacc = 0.0f;

    const float PIH = 1.5707963267948966f;

    for (int l = 0; l < LSEQ; ++l) {
        const size_t row = ((size_t)l * NBATCH + n_batch) * EDIM + colBase;
        if (t < DHEAD) {
            // RoPE: kr = k*cos + rotate_half(k)*sin
            float kd = bf2f(Kbf[row + t]);
            int   dp = (t + 32) & (DHEAD - 1);
            float kp = bf2f(Kbf[row + dp]);
            float rot = (t < 32) ? -kp : kp;
            float kr  = kd * cosT[l * DHEAD + t] + rot * sinT[l * DHEAD + t];
            // cosFormer reweighting
            float idx = PIH * (float)(l + 1) / (float)LSEQ;
            kbuf[t]         = kr * __sinf(idx);
            kbuf[t + DHEAD] = kr * __cosf(idx);
        } else if (t < 2 * DHEAD) {
            vbuf[t - DHEAD] = bf2f(Vbf[row + (t - DHEAD)]);
        }
        __syncthreads();

        float kd = kbuf[d];
        if (half == 0) ksacc += kd;
#pragma unroll
        for (int j = 0; j < 32; ++j) acc[j] += kd * vbuf[m0 + j];
        __syncthreads();
    }

    float* kvOut = kvW + ((size_t)nh * TWOD + d) * DHEAD + m0;
#pragma unroll
    for (int j = 0; j < 32; ++j) kvOut[j] = acc[j];
    if (half == 0) ksumW[(size_t)nh * TWOD + d] = ksacc;
}

// ---------------------------------------------------------------------------
// Per-head output: out[l,m] = (q_[l,:] @ kv[:,m]) / max(q_[l,:]@ksum, eps)
// Grid (NH, L/256); block = 4 groups x 64 threads, kv tile staged in LDS.
// Writes bf16 ctx in (L, N, E) layout for the final WMMA projection.
// ---------------------------------------------------------------------------
__global__ __launch_bounds__(256) void cosformer_out_kernel(
    const unsigned short* __restrict__ Qbf,
    const float* __restrict__ kvW,
    const float* __restrict__ ksumW,
    const float* __restrict__ cosT,
    const float* __restrict__ sinT,
    unsigned short* __restrict__ Ctx)
{
    const int nh      = blockIdx.x;
    const int n_batch = nh / NHEADS;
    const int h       = nh % NHEADS;
    const int colBase = h * DHEAD;

    __shared__ float kvs[TWOD * DHEAD];   // 32 KB
    __shared__ float ksums[TWOD];
    __shared__ float qbuf[4][TWOD];

    const int t = threadIdx.x;
    const int g = t >> 6;          // group 0..3
    const int j = t & 63;          // output column / q element

    for (int i = t; i < TWOD * DHEAD; i += 256) kvs[i] = kvW[(size_t)nh * TWOD * DHEAD + i];
    if (t < TWOD) ksums[t] = ksumW[(size_t)nh * TWOD + t];
    __syncthreads();

    const float PIH   = 1.5707963267948966f;
    const int   lbase = blockIdx.y * 256 + g * 64;

    for (int li = 0; li < 64; ++li) {
        const int l = lbase + li;
        const size_t row = ((size_t)l * NBATCH + n_batch) * EDIM + colBase;

        // RoPE + cosFormer reweighting of this q row (each thread does d = j)
        float qd = bf2f(Qbf[row + j]);
        int   dp = (j + 32) & (DHEAD - 1);
        float qp = bf2f(Qbf[row + dp]);
        float rot = (j < 32) ? -qp : qp;
        float qr  = qd * cosT[l * DHEAD + j] + rot * sinT[l * DHEAD + j];
        float idx = PIH * (float)(l + 1) / (float)LSEQ;
        qbuf[g][j]         = qr * __sinf(idx);
        qbuf[g][j + DHEAD] = qr * __cosf(idx);
        __syncthreads();

        float acc = 0.0f, den = 0.0f;
#pragma unroll 8
        for (int d = 0; d < TWOD; ++d) {
            float qv = qbuf[g][d];
            acc += qv * kvs[d * DHEAD + j];
            den += qv * ksums[d];
        }
        float z = 1.0f / fmaxf(den, 1e-6f);
        Ctx[row + j] = f2bf(acc * z);
        __syncthreads();
    }
}

// ---------------------------------------------------------------------------
extern "C" void kernel_launch(void* const* d_in, const int* in_sizes, int n_in,
                              void* d_out, int out_size, void* d_ws, size_t ws_size,
                              hipStream_t stream) {
    const float* query = (const float*)d_in[0];   // (L, N, E)
    const float* cosT  = (const float*)d_in[1];   // (1, L, D)
    const float* sinT  = (const float*)d_in[2];   // (1, L, D)
    const float* Wq    = (const float*)d_in[3];   // (E, E)
    const float* Wk    = (const float*)d_in[4];
    const float* Wv    = (const float*)d_in[5];
    const float* Wo    = (const float*)d_in[6];
    float* out = (float*)d_out;                   // (L, N, E) fp32

    // ---- workspace carve-up (~89 MB) ----
    char* ws = (char*)d_ws;
    size_t off = 0;
    unsigned short* Xbf   = (unsigned short*)(ws + off); off += (size_t)MROWS * EDIM * 2;
    unsigned short* Wqbf  = (unsigned short*)(ws + off); off += (size_t)EDIM * EDIM * 2;
    unsigned short* Wkbf  = (unsigned short*)(ws + off); off += (size_t)EDIM * EDIM * 2;
    unsigned short* Wvbf  = (unsigned short*)(ws + off); off += (size_t)EDIM * EDIM * 2;
    unsigned short* Wobf  = (unsigned short*)(ws + off); off += (size_t)EDIM * EDIM * 2;
    unsigned short* Qbf   = (unsigned short*)(ws + off); off += (size_t)MROWS * EDIM * 2;
    unsigned short* Kbf   = (unsigned short*)(ws + off); off += (size_t)MROWS * EDIM * 2;
    unsigned short* Vbf   = (unsigned short*)(ws + off); off += (size_t)MROWS * EDIM * 2;
    unsigned short* Ctxbf = (unsigned short*)(ws + off); off += (size_t)MROWS * EDIM * 2;
    float* kvW   = (float*)(ws + off); off += (size_t)NH_TOT * TWOD * DHEAD * 4;
    float* ksumW = (float*)(ws + off); off += (size_t)NH_TOT * TWOD * 4;
    (void)off; (void)ws_size; (void)in_sizes; (void)n_in; (void)out_size;

    // ---- 1. fp32 -> bf16 conversions ----
    const int nX = MROWS * EDIM;       // 8,388,608 (multiple of 1024)
    const int nW = EDIM * EDIM;        // 1,048,576
    f32_to_bf16_kernel<<<nX / 1024, 256, 0, stream>>>(query, Xbf, nX);
    f32_to_bf16_kernel<<<nW / 1024, 256, 0, stream>>>(Wq, Wqbf, nW);
    f32_to_bf16_kernel<<<nW / 1024, 256, 0, stream>>>(Wk, Wkbf, nW);
    f32_to_bf16_kernel<<<nW / 1024, 256, 0, stream>>>(Wv, Wvbf, nW);
    f32_to_bf16_kernel<<<nW / 1024, 256, 0, stream>>>(Wo, Wobf, nW);

    // ---- 2. projections: q = relu(X Wq^T), k = relu(X Wk^T), v = X Wv^T ----
    dim3 gg(MROWS / 128, EDIM / 128);  // (64, 8)
    gemm_wmma_tdm<1><<<gg, 256, 0, stream>>>(Xbf, Wqbf, Qbf, MROWS, EDIM, EDIM);
    gemm_wmma_tdm<1><<<gg, 256, 0, stream>>>(Xbf, Wkbf, Kbf, MROWS, EDIM, EDIM);
    gemm_wmma_tdm<0><<<gg, 256, 0, stream>>>(Xbf, Wvbf, Vbf, MROWS, EDIM, EDIM);

    // ---- 3. linear attention middle ----
    cosformer_kv_kernel<<<NH_TOT, 256, 0, stream>>>(Kbf, Vbf, cosT, sinT, kvW, ksumW);
    dim3 go(NH_TOT, LSEQ / 256);       // (32, 16)
    cosformer_out_kernel<<<go, 256, 0, stream>>>(Qbf, kvW, ksumW, cosT, sinT, Ctxbf);

    // ---- 4. output projection: out = ctx @ Wo^T (fp32 straight to d_out) ----
    gemm_wmma_tdm<2><<<gg, 256, 0, stream>>>(Ctxbf, Wobf, out, MROWS, EDIM, EDIM);
}
